// MoEHTRFeedForward_50208167690490
// MI455X (gfx1250) — matrix operand order repaired and checked by reading, not compiled, not run on hardware
//
#include <hip/hip_runtime.h>
#include <hip/hip_bf16.h>
#include <stdint.h>
#include <stddef.h>

// ---------------- problem constants ----------------
#define N_NODES 8192
#define N_EDGES 262144
#define C_CH    128
#define H_CH    512
#define EC_CH   128
#define M_CO    25
#define NROWS   (N_NODES * M_CO)   // 204800, divisible by 128
#define GCOLS_EQ 2560              // 5 * H

typedef __attribute__((ext_vector_type(16))) __bf16   bf16x16;
typedef __attribute__((ext_vector_type(8)))  float    f32x8;
typedef __attribute__((ext_vector_type(4)))  uint32_t u32x4;

// ---------------- workspace layout (bytes) ----------------
constexpr size_t OFF_SUMS = 0;                                            // [N,128] f32 (becomes c)
constexpr size_t OFF_CNT  = OFF_SUMS + (size_t)N_NODES * EC_CH * 4;       // [N] f32
constexpr size_t OFF_CBF  = OFF_CNT  + (size_t)N_NODES * 4;               // [N,128] bf16
constexpr size_t OFF_W    = OFF_CBF  + (size_t)N_NODES * EC_CH * 2;       // [N,4] f32 router weights
constexpr size_t OFF_W1T  = OFF_W    + (size_t)N_NODES * 4 * 4;           // [4,H,C] bf16 (W1 transposed)
constexpr size_t OFF_W2T  = OFF_W1T  + (size_t)4 * H_CH * C_CH * 2;       // [4,C,H] bf16 (W2 transposed)
constexpr size_t OFF_GWTE = OFF_W2T  + (size_t)4 * C_CH * H_CH * 2;       // [2,2560,128] bf16
constexpr size_t OFF_GWTN = OFF_GWTE + (size_t)2 * GCOLS_EQ * EC_CH * 2;  // [2,512,128] bf16
constexpr size_t OFF_GEQ  = OFF_GWTN + (size_t)2 * H_CH * EC_CH * 2;      // [2,N,2560] bf16 gates*w
constexpr size_t OFF_GNE  = OFF_GEQ  + (size_t)2 * (size_t)N_NODES * GCOLS_EQ * 2; // [2,N,512] bf16

// ---------------- helpers ----------------
__device__ __forceinline__ uint16_t f2bf(float f) {
  uint32_t u = __float_as_uint(f);
  u += 0x7FFFu + ((u >> 16) & 1u);          // round-to-nearest-even
  return (uint16_t)(u >> 16);
}
__device__ __forceinline__ float bf2f(uint16_t h) {
  return __uint_as_float(((uint32_t)h) << 16);
}
// fast SiLU: single v_exp + v_rcp (result is bf16-rounded downstream)
__device__ __forceinline__ float silu(float x) {
  return x * __builtin_amdgcn_rcpf(1.0f + __expf(-x));
}

union FragU { bf16x16 bf; u32x4 q[2]; };

// A operand (16x32 bf16), ISA layout: lanes 0-15 hold K[0..7],K[16..23]; lanes 16-31 K[8..15],K[24..31]
__device__ __forceinline__ bf16x16 load_frag_a(const uint16_t* base, int stride, int lane) {
  int r  = lane & 15;
  int k0 = (lane & 16) ? 8 : 0;
  const uint16_t* q = base + r * stride + k0;
  FragU f;
  f.q[0] = *(const u32x4*)(q);
  f.q[1] = *(const u32x4*)(q + 16);
  return f.bf;
}
// B operand (32x16 bf16) from transposed row-major [N][K] storage:
// lanes 0-15 hold K[0..7],K[16..23] of column lane; lanes 16-31 hold K[8..15],K[24..31]
__device__ __forceinline__ bf16x16 load_frag_b(const uint16_t* base, int stride, int lane) {
  int c  = lane & 15;
  int k0 = (lane & 16) ? 8 : 0;
  const uint16_t* q = base + c * stride + k0;
  FragU f;
  f.q[0] = *(const u32x4*)(q);
  f.q[1] = *(const u32x4*)(q + 16);
  return f.bf;
}
__device__ __forceinline__ f32x8 wmma_bf16(bf16x16 a, bf16x16 b, f32x8 c) {
  return __builtin_amdgcn_wmma_f32_16x16x32_bf16(false, a, false, b, (short)0, c, false, false);
}

// ---------------- small kernels ----------------
__global__ void k_zero(float* p, int n) {
  int i = blockIdx.x * 256 + threadIdx.x;
  if (i < n) p[i] = 0.0f;
}

__global__ void k_scatter(const float* __restrict__ t, const int* __restrict__ dst,
                          float* sums, float* cnt) {
  int i  = blockIdx.x * 256 + threadIdx.x;   // exactly N_EDGES*128 threads
  int e  = i >> 7;
  int ch = i & 127;
  int d  = dst[e];
  unsafeAtomicAdd(&sums[d * EC_CH + ch], t[i]);
  if (ch == 0) unsafeAtomicAdd(&cnt[d], 1.0f);
}

__global__ void k_finalize(float* sums, const float* cnt, uint16_t* cbf) {
  int i = blockIdx.x * 256 + threadIdx.x;    // exactly N*128 threads
  int n = i >> 7;
  float v = sums[i] / fmaxf(cnt[n], 1.0f);
  sums[i] = v;                               // sums now holds c (f32)
  cbf[i]  = f2bf(v);
}

// dst[e][c][r] = src[e][r][c], bf16 convert
__global__ void k_transpose(const float* __restrict__ src, uint16_t* dst, int E, int R, int Cc) {
  int i = blockIdx.x * 256 + threadIdx.x;
  int tot = E * R * Cc;
  if (i >= tot) return;
  int e = i / (R * Cc);
  int rem = i - e * R * Cc;
  int r = rem / Cc;
  int c = rem - r * Cc;
  dst[(size_t)(e * Cc + c) * R + r] = f2bf(src[i]);
}

__global__ void k_router(const float* __restrict__ c, const float* __restrict__ w1,
                         const float* __restrict__ b1, const float* __restrict__ w2,
                         const float* __restrict__ b2, float* wbuf) {
  __shared__ float cs[128];
  __shared__ float hs[128];
  __shared__ float lg[4];
  int n = blockIdx.x, t = threadIdx.x;
  cs[t] = c[n * 128 + t];
  __syncthreads();
  float a = b1[t];
  for (int k = 0; k < 128; k++) a += cs[k] * w1[k * 128 + t];
  hs[t] = silu(a);
  __syncthreads();
  if (t < 4) {
    float L = b2[t];
    for (int k = 0; k < 128; k++) L += hs[k] * w2[k * 4 + t];
    lg[t] = L;
  }
  __syncthreads();
  if (t == 0) {
    float mx = fmaxf(fmaxf(lg[0], lg[1]), fmaxf(lg[2], lg[3]));
    float e0 = __expf(lg[0] - mx), e1 = __expf(lg[1] - mx);
    float e2 = __expf(lg[2] - mx), e3 = __expf(lg[3] - mx);
    float s = __builtin_amdgcn_rcpf(e0 + e1 + e2 + e3);
    wbuf[n * 4 + 0] = e0 * s; wbuf[n * 4 + 1] = e1 * s;
    wbuf[n * 4 + 2] = e2 * s; wbuf[n * 4 + 3] = e3 * s;
  }
}

// G[e][n][col] = silu(c[n,:] @ gw[:,col] + b[col]) * w[n, wofs+e]   (WMMA)
// Two independent accumulator chains in flight.
__global__ void __launch_bounds__(256) k_gates(
    const uint16_t* __restrict__ cbf, const uint16_t* __restrict__ gwt,
    const float* __restrict__ gb, const float* __restrict__ wbuf,
    int wofs, int COLS, uint16_t* G) {
  int lane = threadIdx.x & 31, wid = threadIdx.x >> 5;
  int nt = blockIdx.x * 128, cc = blockIdx.y * 128, e = blockIdx.z;
  const uint16_t* Abase = cbf + (size_t)(nt + 16 * wid) * EC_CH;
  const uint16_t* Bbase = gwt + (size_t)e * COLS * EC_CH + (size_t)cc * EC_CH;
  bf16x16 a[4];
#pragma unroll
  for (int k = 0; k < 4; k++) a[k] = load_frag_a(Abase + k * 32, EC_CH, lane);
  f32x8 z = {0.f, 0.f, 0.f, 0.f, 0.f, 0.f, 0.f, 0.f};
#pragma unroll
  for (int jp = 0; jp < 4; jp++) {           // pairs of 16-col blocks
    f32x8 acc[2] = {z, z};
#pragma unroll
    for (int k = 0; k < 4; k++) {
#pragma unroll
      for (int p = 0; p < 2; p++) {
        int j = 2 * jp + p;
        bf16x16 b = load_frag_b(Bbase + (size_t)(j * 16) * EC_CH + k * 32, EC_CH, lane);
        acc[p] = wmma_bf16(a[k], b, acc[p]);
      }
    }
#pragma unroll
    for (int p = 0; p < 2; p++) {
      int j = 2 * jp + p;
      int col = cc + j * 16 + (lane & 15);
      float bias = gb[e * COLS + col];
#pragma unroll
      for (int r = 0; r < 8; r++) {
        int n = nt + 16 * wid + r + ((lane >> 4) << 3);
        float v = silu(acc[p][r] + bias) * wbuf[n * 4 + wofs + e];
        G[((size_t)(e * N_NODES + n)) * COLS + col] = f2bf(v);
      }
    }
  }
}

// ---------------- main fused expert kernel ----------------
// Per 128-row tile: X (LDS, bf16) loaded once; A-fragments hoisted to registers
// and reused across 4 experts. H processed in 64-col chunks; GEMM1 runs two
// 16-col blocks concurrently (two independent WMMA chains hide B-load latency
// and WMMA->VALU hazard NOPs), gated + stored to the wave-private LDS strip,
// then GEMM2 accumulates into the persistent output accumulators.
__global__ void __launch_bounds__(256) k_main(
    const float* __restrict__ x, const uint16_t* __restrict__ w1t,
    const uint16_t* __restrict__ w2t, const uint16_t* __restrict__ Geq,
    const uint16_t* __restrict__ Gne, const float* __restrict__ neb1,
    const float* __restrict__ neb2, const float* __restrict__ wbuf,
    float* __restrict__ out) {
  __shared__ __align__(16) uint16_t Xt[128 * C_CH];   // 32 KB: X tile bf16
  __shared__ __align__(16) uint16_t Hc[128 * 64];     // 16 KB: gated H chunk (wave-private strips)
  int tid = threadIdx.x, lane = tid & 31, wid = tid >> 5;
  int tile0 = blockIdx.x * 128;

  const float* xsrc = x + (size_t)tile0 * C_CH;
  for (int t = tid; t < 128 * C_CH; t += 256) Xt[t] = f2bf(xsrc[t]);
  if (blockIdx.x + 1 < gridDim.x)
    __builtin_prefetch(x + (size_t)(tile0 + 128) * C_CH + tid * 16, 0, 1);
  __syncthreads();

  // hoist GEMM1 A-fragments (X strip, K=128): invariant across experts/chunks
  const uint16_t* Xw = Xt + (16 * wid) * C_CH;
  bf16x16 ax[4];
#pragma unroll
  for (int k = 0; k < 4; k++) ax[k] = load_frag_a(Xw + k * 32, C_CH, lane);

  f32x8 z = {0.f, 0.f, 0.f, 0.f, 0.f, 0.f, 0.f, 0.f};
  f32x8 outacc[8];
#pragma unroll
  for (int j = 0; j < 8; j++) outacc[j] = z;

  uint16_t* Hw = Hc + (16 * wid) * 64;

  for (int e = 0; e < 4; e++) {
    const uint16_t* W1e = w1t + (size_t)e * H_CH * C_CH;   // [H][C] transposed
    const uint16_t* W2e = w2t + (size_t)e * C_CH * H_CH;   // [C][H] transposed
    for (int hc = 0; hc < 8; hc++) {                        // 64-col H chunks
      // ---- GEMM1: two 16-col blocks in flight, gate, stash to LDS ----
#pragma unroll
      for (int jp = 0; jp < 2; jp++) {
        f32x8 acc[2] = {z, z};
#pragma unroll
        for (int k = 0; k < 4; k++) {
#pragma unroll
          for (int p = 0; p < 2; p++) {
            int j = 2 * jp + p;
            bf16x16 b =
                load_frag_b(W1e + (size_t)(hc * 64 + j * 16) * C_CH + k * 32, C_CH, lane);
            acc[p] = wmma_bf16(ax[k], b, acc[p]);
          }
        }
#pragma unroll
        for (int p = 0; p < 2; p++) {
          int j = 2 * jp + p;
          int hcol = hc * 64 + j * 16 + (lane & 15);
#pragma unroll
          for (int r = 0; r < 8; r++) {
            int lrow = 16 * wid + r + ((lane >> 4) << 3);
            int grow = tile0 + lrow;
            int n = grow / 25;
            int m = grow - n * 25;
            float v = acc[p][r];
            if (e < 2) {
              int l = (m < 1) ? 0 : ((m < 4) ? 1 : ((m < 9) ? 2 : ((m < 16) ? 3 : 4)));
              if (m == 0) v = silu(v);
              v *= bf2f(Geq[((size_t)(e * N_NODES + n)) * GCOLS_EQ + l * H_CH + hcol]);
            } else {
              v = silu(v + neb1[(e - 2) * H_CH + hcol]);
              v *= bf2f(Gne[((size_t)((e - 2) * N_NODES + n)) * H_CH + hcol]);
            }
            Hc[lrow * 64 + j * 16 + (lane & 15)] = f2bf(v);
          }
        }
      }
      // ---- GEMM2 partial-K over this 64-col chunk (LDS in-order per wave) ----
      bf16x16 ah[2];
#pragma unroll
      for (int k = 0; k < 2; k++) ah[k] = load_frag_a(Hw + k * 32, 64, lane);
#pragma unroll
      for (int j = 0; j < 8; j++) {
#pragma unroll
        for (int k = 0; k < 2; k++) {
          bf16x16 b = load_frag_b(W2e + (size_t)(j * 16) * H_CH + hc * 64 + k * 32, H_CH, lane);
          outacc[j] = wmma_bf16(ah[k], b, outacc[j]);
        }
      }
    }
  }
  // ---- epilogue: + sum_ne w[n,e]*b2[e][c], store f32 ----
#pragma unroll
  for (int r = 0; r < 8; r++) {
    int grow = tile0 + 16 * wid + r + ((lane >> 4) << 3);
    int n = grow / 25;
    float w2n = wbuf[n * 4 + 2], w3n = wbuf[n * 4 + 3];
#pragma unroll
    for (int j = 0; j < 8; j++) {
      int ccol = j * 16 + (lane & 15);
      float v = outacc[j][r] + w2n * neb2[ccol] + w3n * neb2[C_CH + ccol];
      out[(size_t)grow * C_CH + ccol] = v;
    }
  }
}

// ---------------- launch ----------------
extern "C" void kernel_launch(void* const* d_in, const int* in_sizes, int n_in,
                              void* d_out, int out_size, void* d_ws, size_t ws_size,
                              hipStream_t stream) {
  (void)in_sizes; (void)n_in; (void)out_size; (void)ws_size;
  const float* x     = (const float*)d_in[0];
  const float* t_ij  = (const float*)d_in[1];
  const int*   eidx  = (const int*)d_in[2];
  const float* eq_w1 = (const float*)d_in[3];
  const float* eq_gw = (const float*)d_in[4];
  const float* eq_gb = (const float*)d_in[5];
  const float* eq_w2 = (const float*)d_in[6];
  const float* ne_w1 = (const float*)d_in[7];
  const float* ne_b1 = (const float*)d_in[8];
  const float* ne_gw = (const float*)d_in[9];
  const float* ne_gb = (const float*)d_in[10];
  const float* ne_w2 = (const float*)d_in[11];
  const float* ne_b2 = (const float*)d_in[12];
  const float* r_w1  = (const float*)d_in[13];
  const float* r_b1  = (const float*)d_in[14];
  const float* r_w2  = (const float*)d_in[15];
  const float* r_b2  = (const float*)d_in[16];

  char* ws = (char*)d_ws;
  float*    sums   = (float*)   (ws + OFF_SUMS);
  float*    cnt    = (float*)   (ws + OFF_CNT);
  uint16_t* cbf    = (uint16_t*)(ws + OFF_CBF);
  float*    wbuf   = (float*)   (ws + OFF_W);
  uint16_t* w1t    = (uint16_t*)(ws + OFF_W1T);
  uint16_t* w2t    = (uint16_t*)(ws + OFF_W2T);
  uint16_t* gwt_eq = (uint16_t*)(ws + OFF_GWTE);
  uint16_t* gwt_ne = (uint16_t*)(ws + OFF_GWTN);
  uint16_t* Geq    = (uint16_t*)(ws + OFF_GEQ);
  uint16_t* Gne    = (uint16_t*)(ws + OFF_GNE);
  float*    outp   = (float*)d_out;

  // 1) zero scatter accumulators (sums + cnt are contiguous)
  {
    int n = N_NODES * EC_CH + N_NODES;
    k_zero<<<(n + 255) / 256, 256, 0, stream>>>(sums, n);
  }
  // 2) scatter-mean numerator/denominator via native f32 atomics
  k_scatter<<<(N_EDGES * EC_CH) / 256, 256, 0, stream>>>(t_ij, eidx + N_EDGES, sums, cnt);
  // 3) c = sums / max(cnt,1)   (f32 in place + bf16 copy)
  k_finalize<<<(N_NODES * EC_CH) / 256, 256, 0, stream>>>(sums, cnt, cbf);
  // 4) weight prep: transpose + bf16 (expert order: eq0,eq1,ne0,ne1)
  k_transpose<<<(2 * 128 * 512 + 255) / 256, 256, 0, stream>>>(eq_w1, w1t, 2, 128, 512);
  k_transpose<<<(2 * 128 * 512 + 255) / 256, 256, 0, stream>>>(ne_w1, w1t + 2 * H_CH * C_CH, 2, 128, 512);
  k_transpose<<<(2 * 512 * 128 + 255) / 256, 256, 0, stream>>>(eq_w2, w2t, 2, 512, 128);
  k_transpose<<<(2 * 512 * 128 + 255) / 256, 256, 0, stream>>>(ne_w2, w2t + 2 * C_CH * H_CH, 2, 512, 128);
  k_transpose<<<(2 * 128 * 2560 + 255) / 256, 256, 0, stream>>>(eq_gw, gwt_eq, 2, 128, 2560);
  k_transpose<<<(2 * 128 * 512 + 255) / 256, 256, 0, stream>>>(ne_gw, gwt_ne, 2, 128, 512);
  // 5) router softmax weights
  k_router<<<N_NODES, 128, 0, stream>>>(sums, r_w1, r_b1, r_w2, r_b2, wbuf);
  // 6) gates (WMMA), with routing weight folded in
  {
    dim3 geq(N_NODES / 128, GCOLS_EQ / 128, 2);
    k_gates<<<geq, 256, 0, stream>>>(cbf, gwt_eq, eq_gb, wbuf, 0, GCOLS_EQ, Geq);
    dim3 gne(N_NODES / 128, H_CH / 128, 2);
    k_gates<<<gne, 256, 0, stream>>>(cbf, gwt_ne, ne_gb, wbuf, 2, H_CH, Gne);
  }
  // 7) fused expert GEMM1 -> gate -> GEMM2 -> mix
  k_main<<<NROWS / 128, 256, 0, stream>>>(x, w1t, w2t, Geq, Gne, ne_b1, ne_b2, wbuf, outp);
}
